// DeformableDetrEncoderLayer_52347061403790
// MI455X (gfx1250) — matrix operand order repaired and checked by reading, not compiled, not run on hardware
//
#include <hip/hip_runtime.h>
#include <hip/hip_bf16.h>

typedef __bf16 bf16_t;
typedef __attribute__((ext_vector_type(16))) __bf16 v16bf;
typedef __attribute__((ext_vector_type(8)))  __bf16 v8bf;
typedef __attribute__((ext_vector_type(8)))  float  v8f;

#define BB   2
#define SS   13294
#define DD   256
#define HH   8
#define PPts 4
#define LLvl 4
#define DHH  32
#define FFNW 1024

__constant__ int c_Hl[4]    = {100, 50, 25, 13};
__constant__ int c_Wl[4]    = {100, 50, 25, 13};
__constant__ int c_start[4] = {0, 10000, 12500, 13125};

// ---------------------------------------------------------------------------
// elementwise helpers
// ---------------------------------------------------------------------------
__global__ void prep_kernel(const float* __restrict__ hs, const float* __restrict__ pos,
                            bf16_t* __restrict__ qbf, bf16_t* __restrict__ hsbf, int n) {
    int i = blockIdx.x * blockDim.x + threadIdx.x;
    if (i < n) {
        float h = hs[i];
        qbf[i]  = (bf16_t)(h + pos[i]);
        hsbf[i] = (bf16_t)h;
    }
}

__global__ void cvt_kernel(const float* __restrict__ x, bf16_t* __restrict__ y, int n) {
    int i = blockIdx.x * blockDim.x + threadIdx.x;
    if (i < n) y[i] = (bf16_t)x[i];
}

// ---------------------------------------------------------------------------
// WMMA GEMM: C[M,N] = A[M,K](bf16 row-major) * B[K,N](bf16 row-major) + bias
// 8 waves / block arranged 2x4; each wave computes a 32x32 register tile
// (2x2 of 16x16 WMMA tiles -> 4 v_wmma per k-step, A/B fragments reused 2x).
// Block tile = 64 x 128.
// ---------------------------------------------------------------------------
__global__ __launch_bounds__(256) void gemm_bf16_wmma(
    const bf16_t* __restrict__ A, const bf16_t* __restrict__ Bw,
    const float* __restrict__ bias,
    float* __restrict__ Cf, bf16_t* __restrict__ Cb,
    int M, int N, int K, int relu)
{
    const int lane  = threadIdx.x & 31;
    const int wv    = threadIdx.x >> 5;
    const int tileM = blockIdx.y * 64  + (wv >> 2) * 32;
    const int tileN = blockIdx.x * 128 + (wv & 3) * 32;
    const int hlf   = lane >> 4;      // K-half selector
    const int l16   = lane & 15;

    int r0 = tileM + l16;       if (r0 >= M) r0 = M - 1;   // clamp loads; stores guarded
    int r1 = tileM + 16 + l16;  if (r1 >= M) r1 = M - 1;

    const bf16_t* a0 = A  + (size_t)r0 * K + hlf * 8;
    const bf16_t* a1 = A  + (size_t)r1 * K + hlf * 8;
    const bf16_t* b0 = Bw + (size_t)(8 * hlf) * N + tileN + l16;
    const bf16_t* b1 = b0 + 16;

    v8f acc00 = {}, acc01 = {}, acc10 = {}, acc11 = {};
    for (int k0 = 0; k0 < K; k0 += 32) {
        union { v16bf v; v8bf h[2]; } fa0, fa1, fb0, fb1;
        const size_t bo = (size_t)k0 * N;
        fa0.h[0] = *(const v8bf*)(a0 + k0);
        fa0.h[1] = *(const v8bf*)(a0 + k0 + 16);
        fa1.h[0] = *(const v8bf*)(a1 + k0);
        fa1.h[1] = *(const v8bf*)(a1 + k0 + 16);
#pragma unroll
        for (int s = 0; s < 8; ++s) {
            fb0.v[s]     = b0[bo + (size_t)s * N];
            fb0.v[8 + s] = b0[bo + (size_t)(16 + s) * N];
            fb1.v[s]     = b1[bo + (size_t)s * N];
            fb1.v[8 + s] = b1[bo + (size_t)(16 + s) * N];
        }
        if (k0 + 32 < K) {
            __builtin_prefetch(a0 + k0 + 32, 0, 3);
            __builtin_prefetch(a1 + k0 + 32, 0, 3);
            __builtin_prefetch(b0 + (size_t)(k0 + 32) * N, 0, 3);
        }
        acc00 = __builtin_amdgcn_wmma_f32_16x16x32_bf16(false, fa0.v, false, fb0.v, (short)0, acc00, false, false);
        acc01 = __builtin_amdgcn_wmma_f32_16x16x32_bf16(false, fa0.v, false, fb1.v, (short)0, acc01, false, false);
        acc10 = __builtin_amdgcn_wmma_f32_16x16x32_bf16(false, fa1.v, false, fb0.v, (short)0, acc10, false, false);
        acc11 = __builtin_amdgcn_wmma_f32_16x16x32_bf16(false, fa1.v, false, fb1.v, (short)0, acc11, false, false);
    }

    // C/D layout: VGPR r, lanes 0-15 -> M=r, lanes 16-31 -> M=r+8; N = l16
    const int c0 = tileN + l16;
    const int c1 = tileN + 16 + l16;
    const float bc0 = bias ? bias[c0] : 0.0f;
    const float bc1 = bias ? bias[c1] : 0.0f;
#pragma unroll
    for (int r = 0; r < 8; ++r) {
        const int rowT = tileM + r + 8 * hlf;        // top 16-row band
        const int rowB = rowT + 16;                  // bottom 16-row band
        if (rowT < M) {
            float v0 = acc00[r] + bc0;
            float v1 = acc01[r] + bc1;
            if (relu) { v0 = fmaxf(v0, 0.0f); v1 = fmaxf(v1, 0.0f); }
            if (Cf) { Cf[(size_t)rowT * N + c0] = v0; Cf[(size_t)rowT * N + c1] = v1; }
            if (Cb) { Cb[(size_t)rowT * N + c0] = (bf16_t)v0; Cb[(size_t)rowT * N + c1] = (bf16_t)v1; }
        }
        if (rowB < M) {
            float v0 = acc10[r] + bc0;
            float v1 = acc11[r] + bc1;
            if (relu) { v0 = fmaxf(v0, 0.0f); v1 = fmaxf(v1, 0.0f); }
            if (Cf) { Cf[(size_t)rowB * N + c0] = v0; Cf[(size_t)rowB * N + c1] = v1; }
            if (Cb) { Cb[(size_t)rowB * N + c0] = (bf16_t)v0; Cb[(size_t)rowB * N + c1] = (bf16_t)v1; }
        }
    }
}

// ---------------------------------------------------------------------------
// softmax over contiguous groups of 16 (the L*P sampling weights per head)
// lane-parallel, width-16 xor shuffles
// ---------------------------------------------------------------------------
__global__ void softmax16_kernel(float* __restrict__ x, int groups) {
    int t = blockIdx.x * blockDim.x + threadIdx.x;
    int g = t >> 4;
    int e = t & 15;
    float xv = (g < groups) ? x[(size_t)g * 16 + e] : 0.0f;
    float mx = xv;
#pragma unroll
    for (int m = 8; m >= 1; m >>= 1) mx = fmaxf(mx, __shfl_xor(mx, m, 16));
    float ex = __expf(xv - mx);
    float sm = ex;
#pragma unroll
    for (int m = 8; m >= 1; m >>= 1) sm += __shfl_xor(sm, m, 16);
    if (g < groups) x[(size_t)g * 16 + e] = ex / sm;
}

// ---------------------------------------------------------------------------
// multi-scale deformable attention sampling:
// one wave per (b, query, head); lane = channel (DH = 32)
// ---------------------------------------------------------------------------
__global__ __launch_bounds__(256) void msda_kernel(
    const float* __restrict__ value,   // (B*S, H*DH)
    const float* __restrict__ offs,    // (B*S, H*L*P*2)
    const float* __restrict__ attw,    // (B*S, H*L*P), softmaxed
    const float* __restrict__ ref,     // (B*S, L, 2)
    bf16_t* __restrict__ out)          // (B*S, H*DH) bf16
{
    int g = blockIdx.x * 8 + (threadIdx.x >> 5);
    if (g >= BB * SS * HH) return;
    const int lane = threadIdx.x & 31;
    const int h  = g & 7;
    const int bq = g >> 3;              // b*S + q
    const int b  = bq / SS;

    const float* refp = ref  + (size_t)bq * (LLvl * 2);
    const float* op   = offs + (size_t)bq * 256 + h * 32;
    const float* ap   = attw + (size_t)bq * 128 + h * 16;

    float acc = 0.0f;
#pragma unroll
    for (int lvl = 0; lvl < LLvl; ++lvl) {
        const int Hh = c_Hl[lvl], Ww = c_Wl[lvl], st = c_start[lvl];
        const float rx = refp[lvl * 2 + 0];
        const float ry = refp[lvl * 2 + 1];
        const float* vlev = value + ((size_t)b * SS + st) * 256 + h * 32 + lane;
#pragma unroll
        for (int p = 0; p < PPts; ++p) {
            float ox = op[lvl * 8 + p * 2 + 0];
            float oy = op[lvl * 8 + p * 2 + 1];
            float x  = (rx + ox / (float)Ww) * (float)Ww - 0.5f;
            float y  = (ry + oy / (float)Hh) * (float)Hh - 0.5f;
            float aw = ap[lvl * 4 + p];

            float x0f = floorf(x), y0f = floorf(y);
            int   x0  = (int)x0f,  y0  = (int)y0f;
            float lx = x - x0f, ly = y - y0f;
            float w00 = (1.0f - lx) * (1.0f - ly);
            float w01 = lx * (1.0f - ly);
            float w10 = (1.0f - lx) * ly;
            float w11 = lx * ly;

            float s = 0.0f;
            // branches are wave-uniform (coords don't depend on lane)
            if (x0 >= 0     && x0 < Ww     && y0 >= 0     && y0 < Hh)
                s += w00 * vlev[(size_t)(y0 * Ww + x0) * 256];
            if (x0 + 1 >= 0 && x0 + 1 < Ww && y0 >= 0     && y0 < Hh)
                s += w01 * vlev[(size_t)(y0 * Ww + x0 + 1) * 256];
            if (x0 >= 0     && x0 < Ww     && y0 + 1 >= 0 && y0 + 1 < Hh)
                s += w10 * vlev[(size_t)((y0 + 1) * Ww + x0) * 256];
            if (x0 + 1 >= 0 && x0 + 1 < Ww && y0 + 1 >= 0 && y0 + 1 < Hh)
                s += w11 * vlev[(size_t)((y0 + 1) * Ww + x0 + 1) * 256];
            acc += aw * s;
        }
    }
    out[(size_t)bq * 256 + h * 32 + lane] = (bf16_t)acc;
}

// ---------------------------------------------------------------------------
// fused residual add + LayerNorm over D=256 (one 256-thread block per row)
// ---------------------------------------------------------------------------
__global__ __launch_bounds__(256) void add_ln_kernel(
    const float* __restrict__ resid, const float* __restrict__ x,
    const float* __restrict__ gam, const float* __restrict__ bet,
    float* __restrict__ outf, bf16_t* __restrict__ outb)
{
    const int row = blockIdx.x;
    const int t   = threadIdx.x;
    const size_t base = (size_t)row * 256 + t;
    float v = resid[base] + x[base];

    float s = v, s2 = v * v;
#pragma unroll
    for (int m = 16; m >= 1; m >>= 1) {
        s  += __shfl_xor(s,  m, 32);
        s2 += __shfl_xor(s2, m, 32);
    }
    __shared__ float ps[8], ps2[8];
    if ((t & 31) == 0) { ps[t >> 5] = s; ps2[t >> 5] = s2; }
    __syncthreads();
    float S = 0.0f, S2 = 0.0f;
#pragma unroll
    for (int i = 0; i < 8; ++i) { S += ps[i]; S2 += ps2[i]; }

    float mu  = S * (1.0f / 256.0f);
    float var = S2 * (1.0f / 256.0f) - mu * mu;
    float r   = rsqrtf(var + 1e-5f);
    float y   = (v - mu) * r * gam[t] + bet[t];
    if (outf) outf[base] = y;
    if (outb) outb[base] = (bf16_t)y;
}

// ---------------------------------------------------------------------------
extern "C" void kernel_launch(void* const* d_in, const int* in_sizes, int n_in,
                              void* d_out, int out_size, void* d_ws, size_t ws_size,
                              hipStream_t stream)
{
    (void)in_sizes; (void)n_in; (void)out_size; (void)ws_size;

    const float* hs    = (const float*)d_in[0];
    const float* pos   = (const float*)d_in[2];
    const float* ref   = (const float*)d_in[3];
    const float* W_off = (const float*)d_in[6];  const float* b_off = (const float*)d_in[7];
    const float* W_att = (const float*)d_in[8];  const float* b_att = (const float*)d_in[9];
    const float* W_val = (const float*)d_in[10]; const float* b_val = (const float*)d_in[11];
    const float* W_out = (const float*)d_in[12]; const float* b_out = (const float*)d_in[13];
    const float* ln1g  = (const float*)d_in[14]; const float* ln1b  = (const float*)d_in[15];
    const float* W_fc1 = (const float*)d_in[16]; const float* b_fc1 = (const float*)d_in[17];
    const float* W_fc2 = (const float*)d_in[18]; const float* b_fc2 = (const float*)d_in[19];
    const float* ln2g  = (const float*)d_in[20]; const float* ln2b  = (const float*)d_in[21];

    const int BS = BB * SS;   // 26588 rows

    char* w = (char*)d_ws;
    auto alloc = [&](size_t bytes) -> char* {
        char* p = w;
        w += (bytes + 255) & ~(size_t)255;
        return p;
    };
    bf16_t* qbf    = (bf16_t*)alloc((size_t)BS * 256 * 2);
    bf16_t* hsbf   = (bf16_t*)alloc((size_t)BS * 256 * 2);
    float*  valf   = (float*) alloc((size_t)BS * 256 * 4);   // value, then reused as GEMM tmp
    float*  offf   = (float*) alloc((size_t)BS * 256 * 4);   // sampling offsets
    float*  attf   = (float*) alloc((size_t)BS * 128 * 4);   // attn logits -> weights
    bf16_t* attnbf = (bf16_t*)alloc((size_t)BS * 256 * 2);   // sampled attention output
    float*  hs1f   = (float*) alloc((size_t)BS * 256 * 4);   // post-LN1 residual stream
    bf16_t* hs1bf  = (bf16_t*)alloc((size_t)BS * 256 * 2);
    bf16_t* actbf  = (bf16_t*)alloc((size_t)BS * 1024 * 2);  // FFN hidden (post-ReLU)
    bf16_t* wvalb  = (bf16_t*)alloc(65536  * 2);
    bf16_t* woffb  = (bf16_t*)alloc(65536  * 2);
    bf16_t* wattb  = (bf16_t*)alloc(32768  * 2);
    bf16_t* woutb  = (bf16_t*)alloc(65536  * 2);
    bf16_t* wfc1b  = (bf16_t*)alloc(262144 * 2);
    bf16_t* wfc2b  = (bf16_t*)alloc(262144 * 2);

    // weights -> bf16
    cvt_kernel<<<(65536  + 255) / 256, 256, 0, stream>>>(W_val, wvalb, 65536);
    cvt_kernel<<<(65536  + 255) / 256, 256, 0, stream>>>(W_off, woffb, 65536);
    cvt_kernel<<<(32768  + 255) / 256, 256, 0, stream>>>(W_att, wattb, 32768);
    cvt_kernel<<<(65536  + 255) / 256, 256, 0, stream>>>(W_out, woutb, 65536);
    cvt_kernel<<<(262144 + 255) / 256, 256, 0, stream>>>(W_fc1, wfc1b, 262144);
    cvt_kernel<<<(262144 + 255) / 256, 256, 0, stream>>>(W_fc2, wfc2b, 262144);

    // q = hs + pos (bf16); hs (bf16)
    prep_kernel<<<(BS * 256 + 255) / 256, 256, 0, stream>>>(hs, pos, qbf, hsbf, BS * 256);

    const dim3 blk(256);
    const int gy = (BS + 63) / 64;   // 64-row block tiles

    // value / offsets / attn-logit projections (WMMA)
    gemm_bf16_wmma<<<dim3(2, gy), blk, 0, stream>>>(hsbf, wvalb, b_val, valf, nullptr, BS, 256,  256, 0);
    gemm_bf16_wmma<<<dim3(2, gy), blk, 0, stream>>>(qbf,  woffb, b_off, offf, nullptr, BS, 256,  256, 0);
    gemm_bf16_wmma<<<dim3(1, gy), blk, 0, stream>>>(qbf,  wattb, b_att, attf, nullptr, BS, 128,  256, 0);

    // softmax over L*P = 16 per (b, q, h)
    softmax16_kernel<<<(BS * 128 + 255) / 256, 256, 0, stream>>>(attf, BS * HH);

    // deformable sampling -> bf16 attention output
    msda_kernel<<<BS, 256, 0, stream>>>(valf, offf, attf, ref, attnbf);

    // output projection (reuse valf as tmp)
    gemm_bf16_wmma<<<dim3(2, gy), blk, 0, stream>>>(attnbf, woutb, b_out, valf, nullptr, BS, 256, 256, 0);

    // LN1(hidden + attn_out)
    add_ln_kernel<<<BS, 256, 0, stream>>>(hs, valf, ln1g, ln1b, hs1f, hs1bf);

    // FFN: fc1 + ReLU (bf16 out), fc2 (f32, reuse valf)
    gemm_bf16_wmma<<<dim3(8, gy), blk, 0, stream>>>(hs1bf, wfc1b, b_fc1, nullptr, actbf, BS, 1024, 256,  1);
    gemm_bf16_wmma<<<dim3(2, gy), blk, 0, stream>>>(actbf, wfc2b, b_fc2, valf,  nullptr, BS, 256,  1024, 0);

    // LN2(hs1 + ffn) -> output
    add_ln_kernel<<<BS, 256, 0, stream>>>(hs1f, valf, ln2g, ln2b, (float*)d_out, nullptr);
}
